// ReModel_Mention_14594298872140
// MI455X (gfx1250) — compile-verified, matrix operand order checked by reading, not run on hardware
//
#include <hip/hip_runtime.h>
#include <hip/hip_bf16.h>

typedef __attribute__((ext_vector_type(8)))  float          v8f;
typedef __attribute__((ext_vector_type(4)))  float          f4;
typedef __attribute__((ext_vector_type(16))) __bf16         v16bf;
typedef __attribute__((ext_vector_type(16))) unsigned short v16us;
typedef __attribute__((ext_vector_type(8)))  unsigned short us8;
typedef __attribute__((ext_vector_type(4)))  unsigned short us4;

#define MINV 1e-10f

__device__ __forceinline__ int imin(int a, int b) { return a < b ? a : b; }

// Native converts: backend uses v_cvt_pk_bf16_f32 / hardware cvt ops.
__device__ __forceinline__ unsigned short f2bf(float x) {
  __bf16 b = (__bf16)x;
  return __builtin_bit_cast(unsigned short, b);
}
__device__ __forceinline__ float bf2f(unsigned short h) {
  return (float)__builtin_bit_cast(__bf16, h);
}

// CDNA5 async copy: global -> LDS, 16 bytes per lane, tracked by ASYNCcnt.
__device__ __forceinline__ void async_b128(unsigned lds_off, const void* gptr) {
  asm volatile("global_load_async_to_lds_b128 %0, %1, off"
               :: "v"(lds_off), "v"((unsigned long long)gptr)
               : "memory");
}
__device__ __forceinline__ void wait_async0() {
  asm volatile("s_wait_asynccnt 0x0" ::: "memory");
}
__device__ __forceinline__ unsigned lds_addr(const void* p) {
  return (unsigned)(unsigned long long)p;   // low 32 bits = LDS offset
}

// Fragment load: 16-bit A/B operand of 16x16x32 WMMA. Per lane the 16
// elements are two contiguous runs of 8 (k = hi*8+e and k = 16+hi*8+e),
// so with k-contiguous LDS rows (80B stride, 16B aligned) this is exactly
// two ds_load_b128.
__device__ __forceinline__ v16bf ldfrag(const unsigned short* row, int hi) {
  us8 x0 = *(const us8*)(row + hi * 8);
  us8 x1 = *(const us8*)(row + 16 + hi * 8);
  v16us u;
#pragma unroll
  for (int e = 0; e < 8; ++e) { u[e] = x0[e]; u[e + 8] = x1[e]; }
  return __builtin_bit_cast(v16bf, u);
}

__device__ __forceinline__ v8f wmma_bf16(v16bf a, v16bf b, v8f c) {
  return __builtin_amdgcn_wmma_f32_16x16x32_bf16(false, a, false, b, (short)0, c,
                                                 false, false);
}

// ---------------------------------------------------------------------------
// Strided-batched GEMM, fp32 in/out, bf16 WMMA compute, fp32 accumulate.
// C[z] (MxN) = A[z/divA] (MxK row-major) * B[z/divB] (KxN, or NxK if transB).
// mode: 0 = store, 1 = C += r, 2 = Cbf = bf16(tanh(C + r + b1[n] + b2[n])).
// Tile MTxNT (8 waves of 32x32, 4 WMMA each per 32-K step), double-buffered
// LDS, clamped M/N staging (OOB rows/cols are never stored), K-tail zeroed.
// ---------------------------------------------------------------------------
template <int MT, int NT>
__global__ __launch_bounds__(256)
void gemm_tpl(const float* __restrict__ A, long long sA, int divA,
              const float* __restrict__ B, long long sB, int divB, int transB,
              float* __restrict__ C, long long sC,
              int M, int N, int K, int mode,
              const float* __restrict__ bias1, const float* __restrict__ bias2,
              unsigned short* __restrict__ Cbf) {
  constexpr int WM = MT / 32;
  __shared__ alignas(16) unsigned short As[2][MT][40];
  __shared__ alignas(16) unsigned short Bs[2][NT][40];
  const int tid = threadIdx.x;
  const float* Ab = A + (long long)((int)blockIdx.z / divA) * sA;
  const float* Bb = B + (long long)((int)blockIdx.z / divB) * sB;
  float* Cb = C + (long long)blockIdx.z * sC;
  const int tm = blockIdx.y * MT, tn = blockIdx.x * NT;
  const int lane = tid & 31, w = tid >> 5;
  const int wm = w % WM, wn = w / WM;
  const int hi = lane >> 4, l16 = lane & 15;
  const bool nfull = (tn + NT <= N);

  auto stage = [&](int buf, int k0) {
    const bool ktail = (k0 + 32 > K);
    // ---- A tile: MT x 32, k-contiguous rows ----
    if (!ktail) {
#pragma unroll
      for (int i = 0; i < MT * 8 / 256; ++i) {
        int idx = tid + i * 256;
        int r = idx >> 3, c4 = (idx & 7) * 4;
        int gr = imin(tm + r, M - 1);                 // clamp: row never stored
        f4 v = *(const f4*)(Ab + (long long)gr * K + k0 + c4);
        us4 s = {f2bf(v.x), f2bf(v.y), f2bf(v.z), f2bf(v.w)};
        *(us4*)&As[buf][r][c4] = s;
      }
    } else {
      for (int idx = tid; idx < MT * 32; idx += 256) {
        int r = idx >> 5, c = idx & 31;
        int gr = imin(tm + r, M - 1), gk = k0 + c;
        float v = (gk < K) ? Ab[(long long)gr * K + gk] : 0.0f;
        As[buf][r][c] = f2bf(v);
      }
    }
    // ---- B tile: stored transposed, NT x 32 k-contiguous rows ----
    if (transB) {                                      // B is [N][K]
      if (!ktail) {
#pragma unroll
        for (int i = 0; i < NT * 8 / 256; ++i) {
          int idx = tid + i * 256;
          int r = idx >> 3, c4 = (idx & 7) * 4;
          int gn = imin(tn + r, N - 1);
          f4 v = *(const f4*)(Bb + (long long)gn * K + k0 + c4);
          us4 s = {f2bf(v.x), f2bf(v.y), f2bf(v.z), f2bf(v.w)};
          *(us4*)&Bs[buf][r][c4] = s;
        }
      } else {
        for (int idx = tid; idx < NT * 32; idx += 256) {
          int r = idx >> 5, c = idx & 31;
          int gn = imin(tn + r, N - 1), gk = k0 + c;
          float v = (gk < K) ? Bb[(long long)gn * K + gk] : 0.0f;
          Bs[buf][r][c] = f2bf(v);
        }
      }
    } else {                                           // B is [K][N]
      if (!ktail && nfull) {
#pragma unroll
        for (int i = 0; i < NT * 8 / 256; ++i) {
          int idx = tid + i * 256;
          int kr = idx / (NT / 4), nc4 = (idx % (NT / 4)) * 4;
          f4 v = *(const f4*)(Bb + (long long)(k0 + kr) * N + tn + nc4);
          Bs[buf][nc4 + 0][kr] = f2bf(v.x);
          Bs[buf][nc4 + 1][kr] = f2bf(v.y);
          Bs[buf][nc4 + 2][kr] = f2bf(v.z);
          Bs[buf][nc4 + 3][kr] = f2bf(v.w);
        }
      } else {
        for (int idx = tid; idx < NT * 32; idx += 256) {
          int kr = idx / NT, nc = idx % NT;
          int gk = k0 + kr, gn = imin(tn + nc, N - 1);
          float v = (gk < K) ? Bb[(long long)gk * N + gn] : 0.0f;
          Bs[buf][nc][kr] = f2bf(v);
        }
      }
    }
  };

  v8f c00 = {}, c01 = {}, c10 = {}, c11 = {};
  stage(0, 0);
  int cur = 0;
  for (int k0 = 0; k0 < K; k0 += 32) {
    __syncthreads();
    v16bf a0 = ldfrag(&As[cur][wm * 32 + l16][0], hi);
    v16bf a1 = ldfrag(&As[cur][wm * 32 + 16 + l16][0], hi);
    v16bf b0 = ldfrag(&Bs[cur][wn * 32 + l16][0], hi);
    v16bf b1 = ldfrag(&Bs[cur][wn * 32 + 16 + l16][0], hi);
    if (k0 + 32 < K) stage(cur ^ 1, k0 + 32);          // overlap with WMMA
    c00 = wmma_bf16(a0, b0, c00);
    c01 = wmma_bf16(a0, b1, c01);
    c10 = wmma_bf16(a1, b0, c10);
    c11 = wmma_bf16(a1, b1, c11);
    cur ^= 1;
  }
  auto st = [&](int m, int n, float val) {
    if (m < M && n < N) {
      long long o = (long long)m * N + n;
      if (mode == 0)      Cb[o] = val;
      else if (mode == 1) Cb[o] += val;
      else Cbf[o] = f2bf(tanhf(Cb[o] + val + bias1[n] + bias2[n]));
    }
  };
#pragma unroll
  for (int r = 0; r < 8; ++r) {                        // C frag: M=r+8*hi
    int m0 = tm + wm * 32 + r + 8 * hi, m1 = m0 + 16;
    int n0 = tn + wn * 32 + l16, n1 = n0 + 16;
    st(m0, n0, c00[r]); st(m0, n1, c01[r]);
    st(m1, n0, c10[r]); st(m1, n1, c11[r]);
  }
}

// ---------------------------------------------------------------------------
// Classifier: out[bp,r] = sum_{n,i,j} h[bp,n,i]*t[bp,n,j]*W[r,n*4096+i*64+j]
// + bias[r].  All three LDS tiles are staged with CDNA5 async copies
// (global_load_async_to_lds_b128, ASYNCcnt): h/t arrive pre-converted to
// bf16 from the dense epilogue, W from a one-time bf16 conversion pass.
// A 32-wide K chunk has fixed (n,i): A frag = h[p,n,i]*t[p,n,j] built as a
// v_pk_mul_bf16 of a splat of h with contiguous us8 loads of the t row.
// Block: 64 pairs x 64 r, 8 waves = 4 ptiles x 2 rtiles, 2 WMMA/wave/step.
// ---------------------------------------------------------------------------
__global__ __launch_bounds__(256)
void classifier_wmma(const unsigned short* __restrict__ hbf,
                     const unsigned short* __restrict__ tbf,
                     const unsigned short* __restrict__ Wb,
                     const float* __restrict__ bc,
                     float* __restrict__ out) {
  __shared__ alignas(16) unsigned short Hs[64][776];   // 1552B row: 16B-mult
  __shared__ alignas(16) unsigned short Ts[64][776];
  __shared__ alignas(16) unsigned short Ws[2][64][40]; // 80B row: 16B-mult
  const int tid = threadIdx.x;
  const int pb = blockIdx.x;                           // 64-pair block (0..63)
  const int rb = blockIdx.y;                           // 64-r block (0..1)
  // async stage h/t tiles: 64 rows x 768 bf16 each, 16B per transfer
  for (int i = tid; i < 64 * 96; i += 256) {
    int r = i / 96, c8 = (i % 96) * 8;
    long long g = (long long)(pb * 64 + r) * 768 + c8;
    async_b128(lds_addr(&Hs[r][c8]), hbf + g);
    async_b128(lds_addr(&Ts[r][c8]), tbf + g);
  }
  const int lane = tid & 31, w = tid >> 5;
  const int pt = w & 3, rt = w >> 2;
  const int hi = lane >> 4, l16 = lane & 15;
  const int prow = pt * 16 + l16;
  const int rrow = rt * 32 + l16;

  auto stageW = [&](int buf, int k0) {                 // 64x32 bf16 = 4KB
    int r = tid >> 2, c8 = (tid & 3) * 8;
    int rg = imin(rb * 64 + r, 96);                    // clamp: never stored
    async_b128(lds_addr(&Ws[buf][r][c8]),
               Wb + (long long)rg * 49152 + k0 + c8);
  };
  stageW(0, 0);
  int cur = 0;
  v8f acc0 = {}, acc1 = {};
  for (int k0 = 0; k0 < 49152; k0 += 32) {
    wait_async0();                                     // staged tiles landed
    __syncthreads();
    v16bf b0 = ldfrag(&Ws[cur][rrow][0], hi);
    v16bf b1 = ldfrag(&Ws[cur][rrow + 16][0], hi);
    const int nb = k0 >> 12;                           // n block
    const int ii = (k0 & 4095) >> 6;                   // i (fixed per chunk)
    const int base = (k0 & 63) + nb * 64;              // j0 + n*64
    const __bf16 hb = __builtin_bit_cast(__bf16, Hs[prow][nb * 64 + ii]);
    us8 t0 = *(const us8*)&Ts[prow][base + hi * 8];
    us8 t1 = *(const us8*)&Ts[prow][base + 16 + hi * 8];
    v16us tu;
    v16bf hvv;
#pragma unroll
    for (int e = 0; e < 8; ++e) { tu[e] = t0[e]; tu[e + 8] = t1[e]; }
#pragma unroll
    for (int e = 0; e < 16; ++e) hvv[e] = hb;
    v16bf av = hvv * __builtin_bit_cast(v16bf, tu);    // v_pk_mul_bf16 x8
    if (k0 + 32 < 49152) stageW(cur ^ 1, k0 + 32);
    acc0 = wmma_bf16(av, b0, acc0);
    acc1 = wmma_bf16(av, b1, acc1);
    cur ^= 1;
  }
#pragma unroll
  for (int r = 0; r < 8; ++r) {
    int pg = pb * 64 + pt * 16 + r + 8 * hi;
    int rg0 = rb * 64 + rt * 32 + l16;
    int rg1 = rg0 + 16;
    if (rg0 < 97) out[(long long)pg * 97 + rg0] = acc0[r] + bc[rg0];
    if (rg1 < 97) out[(long long)pg * 97 + rg1] = acc1[r] + bc[rg1];
  }
}

// ------------------------------ small kernels ------------------------------
__global__ void cvtbf_k(const float* __restrict__ x,
                        unsigned short* __restrict__ y, int n4) {
  int i = blockIdx.x * 256 + threadIdx.x;
  if (i >= n4) return;
  f4 v = *(const f4*)(x + (long long)i * 4);
  us4 s = {f2bf(v.x), f2bf(v.y), f2bf(v.z), f2bf(v.w)};
  *(us4*)(y + (long long)i * 4) = s;
}

__global__ void att_mean_k(const float* __restrict__ att, float* __restrict__ o) {
  int idx = blockIdx.x * 256 + threadIdx.x;            // 1M float4s
  int b = idx >> 18;
  int lm4 = (idx & 262143) * 4;
  const float* p = att + (long long)b * 12 * 1048576 + lm4;
  f4 s = {0.f, 0.f, 0.f, 0.f};
#pragma unroll
  for (int h = 0; h < 12; ++h) s += *(const f4*)(p + (long long)h * 1048576);
  *(f4*)(o + (long long)b * 1048576 + lm4) = s * (1.0f / 12.0f);
}

__global__ void rownorm_k(float* __restrict__ x, int rowlen4) {
  __shared__ float red[256];
  f4* p = (f4*)(x + (long long)blockIdx.x * rowlen4 * 4);
  float s = 0;
  for (int i = threadIdx.x; i < rowlen4; i += 256) {
    f4 v = p[i];
    s += v.x + v.y + v.z + v.w;
  }
  red[threadIdx.x] = s; __syncthreads();
  for (int st = 128; st > 0; st >>= 1) {
    if (threadIdx.x < st) red[threadIdx.x] += red[threadIdx.x + st];
    __syncthreads();
  }
  float inv = 1.0f / (red[0] + MINV);
  for (int i = threadIdx.x; i < rowlen4; i += 256) p[i] *= inv;
}

__global__ void ht_att_k(const int* __restrict__ hts, const float* __restrict__ emap,
                         const float* __restrict__ eatt, float* __restrict__ h_att,
                         float* __restrict__ t_att) {
  int idx = blockIdx.x * 256 + threadIdx.x;            // B*P = 4096
  int b = idx >> 10;
  int hi = hts[idx * 2], ti = hts[idx * 2 + 1];
  float mask = (hi + ti) != 0 ? 1.0f : 0.0f;
  const float* em_b = emap + b * 4200;                 // [42][100]
  const float* ea_b = eatt + b * 4200;
  long long o = (long long)idx * 100;
  float sh = 0, st = 0;
  for (int m = 0; m < 100; ++m) {
    float hm = em_b[hi * 100 + m] * mask;
    float tm = em_b[ti * 100 + m] * mask;
    float ha = ea_b[hi * 100 + m] * tm;
    float ta = ea_b[ti * 100 + m] * hm;
    h_att[o + m] = ha; t_att[o + m] = ta;
    sh += ha; st += ta;
  }
  float ih = 1.0f / (sh + MINV), it = 1.0f / (st + MINV);
  for (int m = 0; m < 100; ++m) { h_att[o + m] *= ih; t_att[o + m] *= it; }
}

__global__ __launch_bounds__(256)
void ctx_att_k(const int* __restrict__ hts, const float* __restrict__ ent_att,
               float* __restrict__ ctx) {
  __shared__ float red[256];
  int pb = blockIdx.x;                                 // (b,p)
  int b = pb >> 10;
  int hidx = hts[pb * 2], tidx = hts[pb * 2 + 1];
  float mask = (hidx + tidx) != 0 ? 1.0f : 0.0f;
  const float* base = ent_att + (long long)b * 12 * 43008;
  int l4 = threadIdx.x * 4;
  f4 s = {0.f, 0.f, 0.f, 0.f};
  for (int h = 0; h < 12; ++h) {
    const float* ph = base + (long long)h * 43008;
    f4 xa = *(const f4*)(ph + hidx * 1024 + l4);
    f4 xb = *(const f4*)(ph + tidx * 1024 + l4);
    s += xa * xb;
  }
  red[threadIdx.x] = s.x + s.y + s.z + s.w; __syncthreads();
  for (int st = 128; st > 0; st >>= 1) {
    if (threadIdx.x < st) red[threadIdx.x] += red[threadIdx.x + st];
    __syncthreads();
  }
  float inv = mask / (red[0] + MINV);                  // mask folded (equiv.)
  *(f4*)(ctx + (long long)pb * 1024 + l4) = s * inv;
}

// ---------------------------------------------------------------------------
extern "C" void kernel_launch(void* const* d_in, const int* in_sizes, int n_in,
                              void* d_out, int out_size, void* d_ws, size_t ws_size,
                              hipStream_t stream) {
  const float* context   = (const float*)d_in[0];
  const float* attention = (const float*)d_in[1];
  const float* mmap      = (const float*)d_in[2];
  const float* emap      = (const float*)d_in[3];
  const int*   hts       = (const int*)d_in[4];
  const float* h_w  = (const float*)d_in[5];
  const float* h_b  = (const float*)d_in[6];
  const float* t_w  = (const float*)d_in[7];
  const float* t_b  = (const float*)d_in[8];
  const float* hc_w = (const float*)d_in[9];
  const float* hc_b = (const float*)d_in[10];
  const float* tc_w = (const float*)d_in[11];
  const float* tc_b = (const float*)d_in[12];
  const float* clas_w = (const float*)d_in[13];
  const float* clas_b = (const float*)d_in[14];
  float* out = (float*)d_out;

  float* W = (float*)d_ws;
  size_t o = 0;
  auto alloc = [&](size_t n) { float* p = W + o; o += n; return p; };
  float* att_mean = alloc(4ull * 1024 * 1024);         // later reused as ctx_att
  float* mention  = alloc(4ull * 100 * 768);
  float* mm_att   = alloc(4ull * 100 * 1024);
  float* ment_att = alloc(4ull * 100 * 100);
  float* eatt_s   = alloc(4ull * 42 * 100);
  float* em       = alloc(4ull * 42 * 1024);
  float* ent_att  = alloc(4ull * 12 * 42 * 1024);
  float* h_att    = alloc(4ull * 1024 * 100);
  float* t_att    = alloc(4ull * 1024 * 100);
  float* hbuf     = alloc(4ull * 1024 * 768);
  float* tbuf     = alloc(4ull * 1024 * 768);
  float* cinfo    = alloc(4ull * 1024 * 768);
  float* lin_h    = alloc(4ull * 1024 * 768);
  float* lin_t    = alloc(4ull * 1024 * 768);
  unsigned short* wbf = (unsigned short*)alloc(2383872);   // bf16 clas_w
  unsigned short* hbf = (unsigned short*)alloc(1572864);   // bf16 tanh(h)
  unsigned short* tbf = (unsigned short*)alloc(1572864);   // bf16 tanh(t)
  float* ctx_att  = att_mean;                          // dead by then -> reuse
  (void)in_sizes; (void)n_in; (void)out_size; (void)ws_size;

  dim3 blk(256);
  const float* np = nullptr;
  unsigned short* nus = nullptr;
  // one-time bf16 conversion of clas_w (97*49152 / 4 = 1191936 float4s)
  cvtbf_k<<<dim3(4656), blk, 0, stream>>>(clas_w, wbf, 1191936);
  att_mean_k<<<dim3(4096), blk, 0, stream>>>(attention, att_mean);
  // mention = mention_map @ context                      [B,100,768]
  gemm_tpl<64,128><<<dim3(6, 2, 4), blk, 0, stream>>>(mmap, 102400, 1, context, 786432, 1, 0, mention, 76800, 100, 768, 1024, 0, np, np, nus);
  // mm_att = mention_map @ att_mean                      [B,100,1024]
  gemm_tpl<64,128><<<dim3(8, 2, 4), blk, 0, stream>>>(mmap, 102400, 1, att_mean, 1048576, 1, 0, mm_att, 102400, 100, 1024, 1024, 0, np, np, nus);
  // mention_att = mm_att @ mention_map^T                 [B,100,100]
  gemm_tpl<64,128><<<dim3(1, 2, 4), blk, 0, stream>>>(mm_att, 102400, 1, mmap, 102400, 1, 1, ment_att, 10000, 100, 100, 1024, 0, np, np, nus);
  // entity_att = entity_map @ mention_att                [B,42,100]
  gemm_tpl<64,128><<<dim3(1, 1, 4), blk, 0, stream>>>(emap, 4200, 1, ment_att, 10000, 1, 0, eatt_s, 4200, 42, 100, 100, 0, np, np, nus);
  // em = normalize_rows(entity_map @ mention_map)        [B,42,1024]
  gemm_tpl<64,128><<<dim3(8, 1, 4), blk, 0, stream>>>(emap, 4200, 1, mmap, 102400, 1, 0, em, 43008, 42, 1024, 100, 0, np, np, nus);
  rownorm_k<<<dim3(168), blk, 0, stream>>>(em, 256);
  // ent_att[b,h] = em[b] @ attention[b,h]                [B,12,42,1024]
  gemm_tpl<64,128><<<dim3(8, 1, 48), blk, 0, stream>>>(em, 43008, 12, attention, 1048576, 1, 0, ent_att, 43008, 42, 1024, 1024, 0, np, np, nus);
  // gathered + normalized h_att / t_att                  [B,P,100]
  ht_att_k<<<dim3(16), blk, 0, stream>>>(hts, emap, eatt_s, h_att, t_att);
  // h = t_att @ mention ; t = h_att @ mention            [B,P,768]
  gemm_tpl<128,64><<<dim3(12, 8, 4), blk, 0, stream>>>(t_att, 102400, 1, mention, 76800, 1, 0, hbuf, 786432, 1024, 768, 100, 0, np, np, nus);
  gemm_tpl<128,64><<<dim3(12, 8, 4), blk, 0, stream>>>(h_att, 102400, 1, mention, 76800, 1, 0, tbuf, 786432, 1024, 768, 100, 0, np, np, nus);
  // ctx_att (gather product over heads, normalize, mask) [B,P,1024]
  ctx_att_k<<<dim3(4096), blk, 0, stream>>>(hts, ent_att, ctx_att);
  // context_info = ctx_att @ context                     [B,P,768]
  gemm_tpl<128,64><<<dim3(12, 8, 4), blk, 0, stream>>>(ctx_att, 1048576, 1, context, 786432, 1, 0, cinfo, 786432, 1024, 768, 1024, 0, np, np, nus);
  // dense pair 1: lin_h = h @ h_w^T ; hbf = bf16(tanh(lin_h + ci @ hc_w^T + b))
  gemm_tpl<128,64><<<dim3(12, 32, 1), blk, 0, stream>>>(hbuf, 0, 1, h_w, 0, 1, 1, lin_h, 0, 4096, 768, 768, 0, np, np, nus);
  gemm_tpl<128,64><<<dim3(12, 32, 1), blk, 0, stream>>>(cinfo, 0, 1, hc_w, 0, 1, 1, lin_h, 0, 4096, 768, 768, 2, h_b, hc_b, hbf);
  // dense pair 2: same for t
  gemm_tpl<128,64><<<dim3(12, 32, 1), blk, 0, stream>>>(tbuf, 0, 1, t_w, 0, 1, 1, lin_t, 0, 4096, 768, 768, 0, np, np, nus);
  gemm_tpl<128,64><<<dim3(12, 32, 1), blk, 0, stream>>>(cinfo, 0, 1, tc_w, 0, 1, 1, lin_t, 0, 4096, 768, 768, 2, t_b, tc_b, tbf);
  // group-biLinear classifier (fused, async-staged, no 800MB bl tensor)
  classifier_wmma<<<dim3(64, 2), blk, 0, stream>>>(hbf, tbf, wbf, clas_b, out);
}
